// SparseConv2D_58188216926912
// MI455X (gfx1250) — matrix operand order repaired and checked by reading, not compiled, not run on hardware
//
#include <hip/hip_runtime.h>

// ---- CDNA5 wave32 WMMA types -------------------------------------------------
typedef __bf16 v16bf __attribute__((ext_vector_type(16)));
typedef __bf16 v8bf  __attribute__((ext_vector_type(8)));
typedef __bf16 v4bf  __attribute__((ext_vector_type(4)));
typedef float  v8f   __attribute__((ext_vector_type(8)));

#define Fc   512
#define Cc   512
#define Pc   65536            // H*W = 256*256
#define KT   32               // K-step (bf16 WMMA K)
#define LDSS 40               // padded row stride in bf16 elems (80B rows, 16B aligned)

// A-fragment (16x32 bf16, row-major LDS tile [row][c], stride LDSS):
// lane L: M = L&15, h = L>>4; elems 0..7 = K(8h..8h+7), elems 8..15 = K(16+8h..16+8h+7)
__device__ __forceinline__ v16bf frag_a(const __bf16* base, int row, int h) {
  const __bf16* p = base + row * LDSS + 8 * h;
  v8bf lo = *(const v8bf*)(p);
  v8bf hi = *(const v8bf*)(p + 16);
  v16bf r;
#pragma unroll
  for (int i = 0; i < 8; ++i) { r[i] = lo[i]; r[i + 8] = hi[i]; }
  return r;
}

// B-fragment (32x16 bf16, LDS tile transposed [p][c], stride LDSS):
// lane L: N = L&15, h = L>>4; elems 0..7 = K(8h..8h+7), elems 8..15 = K(16+8h..16+8h+7)
__device__ __forceinline__ v16bf frag_b(const __bf16* base, int col, int h) {
  const __bf16* p = base + col * LDSS + 8 * h;
  v8bf lo = *(const v8bf*)(p);
  v8bf hi = *(const v8bf*)(p + 16);
  v16bf r;
#pragma unroll
  for (int i = 0; i < 8; ++i) { r[i] = lo[i]; r[i + 8] = hi[i]; }
  return r;
}

struct bf2 { __bf16 hi, lo; };
__device__ __forceinline__ bf2 split_bf16(float x) {
  bf2 r;
  r.hi = (__bf16)x;
  r.lo = (__bf16)(x - (float)r.hi);
  return r;
}

__global__ __launch_bounds__(256)
void spconv1x1_splitbf16_wmma(const float* __restrict__ X,     // [C][P]
                              const float* __restrict__ Kd,    // [F][C] (already masked)
                              const float* __restrict__ bias,  // [F]
                              float* __restrict__ out)         // [F][P]
{
  __shared__ __bf16 sAh[128 * LDSS];
  __shared__ __bf16 sAl[128 * LDSS];
  __shared__ __bf16 sBh[128 * LDSS];   // transposed: [p][c]
  __shared__ __bf16 sBl[128 * LDSS];

  const int tid  = threadIdx.x;
  const int lane = tid & 31;
  const int wid  = tid >> 5;
  const int wm   = wid >> 2;          // 0..1  (64-row F strip)
  const int wn   = wid & 3;           // 0..3  (32-col P strip)
  const int h    = lane >> 4;
  const int l16  = lane & 15;

  const int fBase = blockIdx.y * 128;
  const int pBase = blockIdx.x * 128;

  // A-tile loader coords: row = (pass*32 + tid/8), 4 consecutive c from q*4
  const int aRow = tid >> 3;
  const int aQ   = tid & 7;
  // B-tile loader coords: p = tid&127, c0 = pass*8 + (tid>>7)*4
  const int bP   = tid & 127;
  const int bC   = (tid >> 7) * 4;

  v8f acc[4][2];
#pragma unroll
  for (int mt = 0; mt < 4; ++mt)
#pragma unroll
    for (int nt = 0; nt < 2; ++nt)
      acc[mt][nt] = (v8f){};

  for (int k0 = 0; k0 < Cc; k0 += KT) {
    __syncthreads();   // previous compute phase done reading LDS

    // ---- stage K-tile of the (dense-stored sparse) kernel: 128F x 32C ----
#pragma unroll
    for (int pass = 0; pass < 4; ++pass) {
      const int row = pass * 32 + aRow;                 // 0..127
      float4 v = *(const float4*)(Kd + (fBase + row) * Cc + k0 + aQ * 4);
      v4bf hi4, lo4;
      bf2 s0 = split_bf16(v.x); hi4[0] = s0.hi; lo4[0] = s0.lo;
      bf2 s1 = split_bf16(v.y); hi4[1] = s1.hi; lo4[1] = s1.lo;
      bf2 s2 = split_bf16(v.z); hi4[2] = s2.hi; lo4[2] = s2.lo;
      bf2 s3 = split_bf16(v.w); hi4[3] = s3.hi; lo4[3] = s3.lo;
      *(v4bf*)(&sAh[row * LDSS + aQ * 4]) = hi4;
      *(v4bf*)(&sAl[row * LDSS + aQ * 4]) = lo4;
    }

    // ---- stage X-tile transposed: 32C x 128P -> LDS[p][c] ----
#pragma unroll
    for (int pass = 0; pass < 4; ++pass) {
      const int c0 = pass * 8 + bC;                     // 0..31
      v4bf hi4, lo4;
#pragma unroll
      for (int i = 0; i < 4; ++i) {
        float x = X[(k0 + c0 + i) * Pc + pBase + bP];   // lane-coalesced along p
        bf2 s = split_bf16(x);
        hi4[i] = s.hi;
        lo4[i] = s.lo;
      }
      *(v4bf*)(&sBh[bP * LDSS + c0]) = hi4;
      *(v4bf*)(&sBl[bP * LDSS + c0]) = lo4;
    }

    __syncthreads();   // tiles visible

    // ---- prefetch next K-tile lines while we compute (global_prefetch_b8) ----
    if (k0 + KT < Cc) {
      __builtin_prefetch(Kd + (fBase + aRow) * Cc + (k0 + KT) + aQ * 4, 0, 1);
      __builtin_prefetch(X + (k0 + KT + bC) * Pc + pBase + bP, 0, 1);
    }

    // ---- fragments ----
    v16bf Ah[4], Al[4], Bh[2], Bl[2];
#pragma unroll
    for (int mt = 0; mt < 4; ++mt) {
      const int row = wm * 64 + mt * 16 + l16;
      Ah[mt] = frag_a(sAh, row, h);
      Al[mt] = frag_a(sAl, row, h);
    }
#pragma unroll
    for (int nt = 0; nt < 2; ++nt) {
      const int col = wn * 32 + nt * 16 + l16;
      Bh[nt] = frag_b(sBh, col, h);
      Bl[nt] = frag_b(sBl, col, h);
    }

    // ---- split-bf16 GEMM: Kh*Xh + Kh*Xl + Kl*Xh (fp32 accumulate) ----
#pragma unroll
    for (int mt = 0; mt < 4; ++mt) {
#pragma unroll
      for (int nt = 0; nt < 2; ++nt) {
        acc[mt][nt] = __builtin_amdgcn_wmma_f32_16x16x32_bf16(
            false, Ah[mt], false, Bh[nt], (short)0, acc[mt][nt], false, false);
        acc[mt][nt] = __builtin_amdgcn_wmma_f32_16x16x32_bf16(
            false, Ah[mt], false, Bl[nt], (short)0, acc[mt][nt], false, false);
        acc[mt][nt] = __builtin_amdgcn_wmma_f32_16x16x32_bf16(
            false, Al[mt], false, Bh[nt], (short)0, acc[mt][nt], false, false);
      }
    }
  }

  // ---- epilogue: bias + ReLU, fused store ----
  // C/D layout: VGPR r, lane L -> M = 8*h + r, N = L&15
#pragma unroll
  for (int mt = 0; mt < 4; ++mt) {
    const int f0 = fBase + wm * 64 + mt * 16 + 8 * h;
#pragma unroll
    for (int nt = 0; nt < 2; ++nt) {
      const int p = pBase + wn * 32 + nt * 16 + l16;
#pragma unroll
      for (int r = 0; r < 8; ++r) {
        const int f = f0 + r;
        float o = acc[mt][nt][r] + bias[f];
        out[f * Pc + p] = fmaxf(o, 0.0f);
      }
    }
  }
}

extern "C" void kernel_launch(void* const* d_in, const int* in_sizes, int n_in,
                              void* d_out, int out_size, void* d_ws, size_t ws_size,
                              hipStream_t stream) {
  (void)in_sizes; (void)n_in; (void)out_size; (void)d_ws; (void)ws_size;
  const float* X    = (const float*)d_in[0];   // inputs [1,C,H,W] fp32
  const float* Kd   = (const float*)d_in[1];   // kernel [F,C] fp32 (mask pre-applied in setup)
  // d_in[2] = mask: redundant (kernel already masked) -> ignored
  const float* bias = (const float*)d_in[3];   // [F] fp32
  float* out = (float*)d_out;                  // [1,F,H,W] fp32

  dim3 grid(Pc / 128, Fc / 128, 1);            // 512 x 4 = 2048 workgroups
  spconv1x1_splitbf16_wmma<<<grid, 256, 0, stream>>>(X, Kd, bias, out);
}
